// LiquidMoE_17300128268516
// MI455X (gfx1250) — matrix operand order repaired and checked by reading, not compiled, not run on hardware
//
#include <hip/hip_runtime.h>
#include <math.h>

// Problem constants (from reference): B=2, S=4096, D=1024, E=16, F=4096, TOPK=4
#define T_TOK 8192
#define DDIM  1024
#define EEXP  16
#define FDIM  4096
#define KTOP  4
#define FCHUNK 2048            // F processed in 2 chunks to shrink LDS

typedef __bf16 bf16_t;
typedef __attribute__((ext_vector_type(16))) bf16_t v16bf;
typedef __attribute__((ext_vector_type(8)))  bf16_t v8bf;
typedef __attribute__((ext_vector_type(8)))  float  v8f;
typedef __attribute__((ext_vector_type(4)))  float  v4f;

// LDS strides in bf16 elements; +8 keeps 16B alignment and rotates banks
// (row stride in dwords % 64 == 4 -> 16 rows land on distinct bank groups).
#define A_STR (DDIM + 8)       // 1032 halves = 2064 B per row
#define H_STR (FCHUNK + 8)     // 2056 halves = 4112 B per row

// ---------------------------------------------------------------------------
// Kernel 0: fused fp32->bf16 convert + transpose:  out[e][c][r] = in[e][r][c].
// LDS 32x33 tile keeps both global sides coalesced. One-time pass that
// removes all converts / strided gathers from the WMMA hot loop and halves
// steady-state weight traffic (bf16 vs fp32).
// ---------------------------------------------------------------------------
__global__ __launch_bounds__(256) void transpose_cvt_bf16(
    const float* __restrict__ in, bf16_t* __restrict__ outp, int R, int C)
{
    __shared__ bf16_t tile[32][33];
    const int e = blockIdx.z;
    const float* src = in + (size_t)e * R * C;
    bf16_t*      dst = outp + (size_t)e * R * C;
    const int c0 = blockIdx.x * 32;
    const int r0 = blockIdx.y * 32;
    const int tx = threadIdx.x;          // 0..31
    const int ty = threadIdx.y;          // 0..7
#pragma unroll
    for (int i = 0; i < 32; i += 8)
        tile[ty + i][tx] = (bf16_t)src[(size_t)(r0 + ty + i) * C + c0 + tx];
    __syncthreads();
#pragma unroll
    for (int i = 0; i < 32; i += 8)
        dst[(size_t)(c0 + ty + i) * R + r0 + tx] = tile[tx][ty + i];
}

// ---------------------------------------------------------------------------
// Kernel 1: gating. One wave32 per token. gates = x @ gate_w^T, scaled by
// sigmoid(trust); top-4 (lowest index wins ties, matching jax.lax.top_k),
// softmax over the 4 values.
// ---------------------------------------------------------------------------
__global__ __launch_bounds__(256) void moe_gate_topk(
    const float* __restrict__ x, const float* __restrict__ gw,
    const float* __restrict__ trust,
    int* __restrict__ tki, float* __restrict__ tkp)
{
    const int lane = threadIdx.x & 31;
    const int wave = threadIdx.x >> 5;
    const int t = blockIdx.x * 8 + wave;

    const float* xr = x + (size_t)t * DDIM;
    float xv[32];
#pragma unroll
    for (int i = 0; i < 32; ++i) xv[i] = xr[lane + i * 32];

    float g[EEXP];
#pragma unroll
    for (int e = 0; e < EEXP; ++e) {
        const float* wr = gw + (size_t)e * DDIM;
        float acc = 0.f;
#pragma unroll
        for (int i = 0; i < 32; ++i) acc += xv[i] * wr[lane + i * 32];
#pragma unroll
        for (int off = 16; off; off >>= 1) acc += __shfl_xor(acc, off, 32);
        const float ts = trust[e];
        g[e] = acc * (1.0f / (1.0f + expf(-ts)));   // trust-weighted gate
    }

    if (lane == 0) {
        int   idx[KTOP];
        float val[KTOP];
        unsigned taken = 0u;
#pragma unroll
        for (int k = 0; k < KTOP; ++k) {
            int best = 0; float bv = -3.4e38f;
#pragma unroll
            for (int e = 0; e < EEXP; ++e)
                if (!((taken >> e) & 1u) && g[e] > bv) { bv = g[e]; best = e; }
            taken |= 1u << best;
            idx[k] = best; val[k] = bv;
        }
        float p[KTOP], s = 0.f;
#pragma unroll
        for (int k = 0; k < KTOP; ++k) { p[k] = expf(val[k] - val[0]); s += p[k]; }
        const float inv = 1.0f / s;
#pragma unroll
        for (int k = 0; k < KTOP; ++k) {
            tki[t * KTOP + k] = idx[k];
            tkp[t * KTOP + k] = p[k] * inv;
        }
    }
}

// ---------------------------------------------------------------------------
// Kernel 2: scatter routed (token, prob) pairs into per-expert buckets.
// ---------------------------------------------------------------------------
__global__ __launch_bounds__(256) void moe_scatter(
    const int* __restrict__ tki, const float* __restrict__ tkp,
    int* __restrict__ cnt, int* __restrict__ bucket, float* __restrict__ bprob)
{
    const int t = blockIdx.x * 256 + threadIdx.x;
#pragma unroll
    for (int k = 0; k < KTOP; ++k) {
        const int e = tki[t * KTOP + k];
        const int pos = atomicAdd(&cnt[e], 1);
        bucket[e * T_TOK + pos] = t;
        bprob[e * T_TOK + pos] = tkp[t * KTOP + k];
    }
}

// ---------------------------------------------------------------------------
// Kernel 3: fused expert FFN. Block = (16-token tile, expert). 256 threads
// = 8 wave32s. F is processed in 2 chunks of 2048: phase-2 fills a 66 KB
// H-chunk in LDS, phase-3 folds it into persistent per-wave Y accumulators
// (8 N-tiles x v8f = 64 VGPRs/lane). Total LDS ~99 KB -> 3 blocks/WGP
// (24 waves) for latency hiding, vs 1 block with a monolithic H tile.
// Weights are pre-transposed bf16 (w1t=[E][F][D], w2t=[E][D][F]) so every
// B fragment is two contiguous global_load_b128 per lane, no converts.
// Two N-tiles per K-iteration share one A fragment (2 WMMAs per A-load).
// ---------------------------------------------------------------------------
__global__ __launch_bounds__(256) void moe_ffn_wmma(
    const float* __restrict__ x,
    const bf16_t* __restrict__ w1t, const float* __restrict__ b1,
    const bf16_t* __restrict__ w2t, const float* __restrict__ b2,
    const int* __restrict__ cnt, const int* __restrict__ bucket,
    const float* __restrict__ bprob, float* __restrict__ out)
{
    extern __shared__ __align__(16) char smem[];
    bf16_t* sA   = (bf16_t*)smem;                // [16][A_STR] bf16 : 33,024 B
    bf16_t* sH   = sA + 16 * A_STR;              // [16][H_STR] bf16 : 65,792 B
    int*    sTok = (int*)(sH + 16 * H_STR);      // [16]
    float*  sPrb = (float*)(sTok + 16);          // [16]

    const int e  = blockIdx.y;
    const int m0 = blockIdx.x * 16;
    const int nrow = cnt[e];
    if (m0 >= nrow) return;                      // block-uniform: EXEC stays full

    const bf16_t* w1te = w1t + (size_t)e * DDIM * FDIM;   // [F][D]
    const bf16_t* w2te = w2t + (size_t)e * FDIM * DDIM;   // [D][F]
    const float*  b1e  = b1 + (size_t)e * FDIM;
    const float*  b2e  = b2 + (size_t)e * DDIM;

    const int tid = threadIdx.x;

    // Phase 0: row metadata (pad rows beyond count with token 0, prob 0)
    if (tid < 16) {
        const int r = m0 + tid;
        int tok = 0; float p = 0.f;
        if (r < nrow) { tok = bucket[e * T_TOK + r]; p = bprob[e * T_TOK + r]; }
        sTok[tid] = tok; sPrb[tid] = p;
    }
    __syncthreads();

    // Phase 1: gather 16 x rows, fp32 -> bf16, into LDS A tile
    {
        const int row = tid >> 4;            // 0..15
        const int cb  = (tid & 15) * 64;     // 64 columns per thread
        const float* xr = x + (size_t)sTok[row] * DDIM + cb;
        bf16_t* dst = sA + row * A_STR + cb;
#pragma unroll
        for (int j = 0; j < 8; ++j) {
            const v4f a = *(const v4f*)(xr + j * 8);
            const v4f b = *(const v4f*)(xr + j * 8 + 4);
            v8bf hv;
#pragma unroll
            for (int q = 0; q < 4; ++q) { hv[q] = (bf16_t)a[q]; hv[q + 4] = (bf16_t)b[q]; }
            *(v8bf*)(dst + j * 8) = hv;
        }
    }
    __syncthreads();

    const int lane = tid & 31;
    const int wv   = tid >> 5;      // wave id 0..7
    const int nl   = lane & 15;     // N / M-row index within 16
    const int hi   = lane >> 4;     // K-half selector per ISA fragment layout
    union AU { v16bf v; v8bf h[2]; };

    // A-fragment base in sA: M=nl; lanes<16 take K {0..7,16..23}+k0,
    // lanes>=16 take {8..15,24..31}+k0.  B-fragment: same K split of
    // column n -> two 16B pieces in the transposed bf16 weight image.
    const bf16_t* apA = sA + nl * A_STR + hi * 8;
    const bf16_t* apH = sH + nl * H_STR + hi * 8;

    // Persistent second-GEMM accumulators: wave wv owns D cols [wv*128,+128)
    v8f yacc[8];
#pragma unroll
    for (int i = 0; i < 8; ++i) yacc[i] = (v8f){};

    for (int c = 0; c < FDIM / FCHUNK; ++c) {
        const int f0 = c * FCHUNK;

        // ---- Phase 2: sH = gelu(A @ W1[:, f0:f0+FCHUNK] + b1) ----
        // wave wv owns local F cols [wv*256, +256) = 16 N-tiles (8 pairs)
        for (int nt = 0; nt < 16; nt += 2) {
            const int nloc = wv * 256 + nt * 16;         // chunk-local column
            const int nglb = f0 + nloc;                  // global F column
            v8f acc0 = {}, acc1 = {};
            const bf16_t* wp0 = w1te + (size_t)(nglb + nl) * DDIM + hi * 8;
            const bf16_t* wp1 = wp0 + (size_t)16 * DDIM;
            for (int k0 = 0; k0 < DDIM; k0 += 32) {
                AU au;
                au.h[0] = *(const v8bf*)(apA + k0);
                au.h[1] = *(const v8bf*)(apA + k0 + 16);
                AU bu0, bu1;
                bu0.h[0] = *(const v8bf*)(wp0 + k0);
                bu0.h[1] = *(const v8bf*)(wp0 + k0 + 16);
                bu1.h[0] = *(const v8bf*)(wp1 + k0);
                bu1.h[1] = *(const v8bf*)(wp1 + k0 + 16);
                acc0 = __builtin_amdgcn_wmma_f32_16x16x32_bf16(
                    false, au.v, false, bu0.v, (short)0, acc0, false, false);
                acc1 = __builtin_amdgcn_wmma_f32_16x16x32_bf16(
                    false, au.v, false, bu1.v, (short)0, acc1, false, false);
            }
            const float bias0 = b1e[nglb + nl];
            const float bias1 = b1e[nglb + 16 + nl];
            bf16_t* h0 = sH + nloc + nl;
            bf16_t* h1 = h0 + 16;
#pragma unroll
            for (int i = 0; i < 8; ++i) {    // C layout: M = hi*8 + i, N = nl
                float v0 = acc0[i] + bias0;
                float v1 = acc1[i] + bias1;
                v0 = 0.5f * v0 * (1.0f + erff(v0 * 0.70710678118654752f));
                v1 = 0.5f * v1 * (1.0f + erff(v1 * 0.70710678118654752f));
                h0[(hi * 8 + i) * H_STR] = (bf16_t)v0;
                h1[(hi * 8 + i) * H_STR] = (bf16_t)v1;
            }
        }
        __syncthreads();   // sH chunk complete before any wave consumes it

        // ---- Phase 3 (partial): yacc += sH-chunk @ W2[f0:f0+FCHUNK, :] ----
        for (int nt = 0; nt < 8; nt += 2) {
            const int n0 = wv * 128 + nt * 16;
            const bf16_t* wp0 = w2te + (size_t)(n0 + nl) * FDIM + f0 + hi * 8;
            const bf16_t* wp1 = wp0 + (size_t)16 * FDIM;
            v8f acc0 = yacc[nt], acc1 = yacc[nt + 1];
            for (int k0 = 0; k0 < FCHUNK; k0 += 32) {
                AU au;
                au.h[0] = *(const v8bf*)(apH + k0);
                au.h[1] = *(const v8bf*)(apH + k0 + 16);
                AU bu0, bu1;
                bu0.h[0] = *(const v8bf*)(wp0 + k0);
                bu0.h[1] = *(const v8bf*)(wp0 + k0 + 16);
                bu1.h[0] = *(const v8bf*)(wp1 + k0);
                bu1.h[1] = *(const v8bf*)(wp1 + k0 + 16);
                acc0 = __builtin_amdgcn_wmma_f32_16x16x32_bf16(
                    false, au.v, false, bu0.v, (short)0, acc0, false, false);
                acc1 = __builtin_amdgcn_wmma_f32_16x16x32_bf16(
                    false, au.v, false, bu1.v, (short)0, acc1, false, false);
            }
            yacc[nt] = acc0; yacc[nt + 1] = acc1;
        }
        __syncthreads();   // chunk consumed before next phase-2 overwrites sH
    }

    // ---- Epilogue: bias, routing-prob scale, atomic combine into out ----
#pragma unroll
    for (int nt = 0; nt < 8; ++nt) {
        const int n0 = wv * 128 + nt * 16;
        const float bias = b2e[n0 + nl];
#pragma unroll
        for (int i = 0; i < 8; ++i) {
            const int m = hi * 8 + i;
            const float yv = (yacc[nt][i] + bias) * sPrb[m];
            atomicAdd(out + (size_t)sTok[m] * DDIM + n0 + nl, yv);
        }
    }
}

// ---------------------------------------------------------------------------
extern "C" void kernel_launch(void* const* d_in, const int* in_sizes, int n_in,
                              void* d_out, int out_size, void* d_ws, size_t ws_size,
                              hipStream_t stream)
{
    (void)in_sizes; (void)n_in; (void)out_size; (void)ws_size;

    const float* x      = (const float*)d_in[0];
    const float* gate_w = (const float*)d_in[1];
    const float* trust  = (const float*)d_in[2];
    const float* w1     = (const float*)d_in[3];
    const float* b1     = (const float*)d_in[4];
    const float* w2     = (const float*)d_in[5];
    const float* b2     = (const float*)d_in[6];
    float* out = (float*)d_out;

    // Workspace carve: ~1.3 MB routing + 2 x 128 MB bf16 transposed weights.
    char* ws = (char*)d_ws;
    int*    cnt    = (int*)ws;    ws += 256;
    int*    tki    = (int*)ws;    ws += (size_t)T_TOK * KTOP * sizeof(int);
    float*  tkp    = (float*)ws;  ws += (size_t)T_TOK * KTOP * sizeof(float);
    int*    bucket = (int*)ws;    ws += (size_t)EEXP * T_TOK * sizeof(int);
    float*  bprob  = (float*)ws;  ws += (size_t)EEXP * T_TOK * sizeof(float);
    ws = (char*)(((uintptr_t)ws + 255) & ~(uintptr_t)255);
    bf16_t* w1t    = (bf16_t*)ws; ws += (size_t)EEXP * DDIM * FDIM * sizeof(bf16_t);
    bf16_t* w2t    = (bf16_t*)ws;

    hipMemsetAsync(out, 0, (size_t)T_TOK * DDIM * sizeof(float), stream);
    hipMemsetAsync(cnt, 0, EEXP * sizeof(int), stream);

    // One-time weight conditioning: fp32 -> bf16 + transpose (coalesced).
    {
        dim3 blk(32, 8);
        dim3 g1(FDIM / 32, DDIM / 32, EEXP);   // w1 [E][D][F] -> w1t [E][F][D]
        transpose_cvt_bf16<<<g1, blk, 0, stream>>>(w1, w1t, DDIM, FDIM);
        dim3 g2(DDIM / 32, FDIM / 32, EEXP);   // w2 [E][F][D] -> w2t [E][D][F]
        transpose_cvt_bf16<<<g2, blk, 0, stream>>>(w2, w2t, FDIM, DDIM);
    }

    moe_gate_topk<<<T_TOK / 8, 256, 0, stream>>>(x, gate_w, trust, tki, tkp);
    moe_scatter<<<T_TOK / 256, 256, 0, stream>>>(tki, tkp, cnt, bucket, bprob);

    dim3 grid(T_TOK / 16, EEXP);   // worst case: every token on one expert
    const size_t lds = (size_t)(16 * A_STR + 16 * H_STR) * sizeof(bf16_t)
                     + 16 * sizeof(int) + 16 * sizeof(float);
    moe_ffn_wmma<<<grid, 256, lds, stream>>>(x, w1t, b1, w2t, b2,
                                             cnt, bucket, bprob, out);
}
